// Int4Linear_73778948211032
// MI455X (gfx1250) — matrix operand order repaired and checked by reading, not compiled, not run on hardware
//
#include <hip/hip_runtime.h>

// ------------------------------------------------------------------
// Int4 group-quantized linear  Y[M,N] = X[M,K] * dequant(Wp)[K,N] + bias
//   Wp: [K/2, N] int32, low nibble = even k, high nibble = odd k
//   scales/zeros: [K/128, N]
// Tile 128x128x128, dequant int4 -> f16 via mantissa bit-trick +
// v_pk_fma_f16, stage in LDS, v_wmma_f32_16x16x32_f16, f32 accum,
// fused bias, non-temporal output stores (preserve X/W in 192MB L2).
// ------------------------------------------------------------------

#define BM 128
#define BN 128
#define BK 128
#define GS 128
#define LDA (BK + 8)   // padded LDS stride (halves): 272B rows -> conflict-free b128 frags

typedef __attribute__((ext_vector_type(16))) _Float16 v16h;
typedef __attribute__((ext_vector_type(8)))  _Float16 v8h;
typedef __attribute__((ext_vector_type(2)))  _Float16 h2;
typedef __attribute__((ext_vector_type(8)))  float    v8f;

// A-matrix 16x32 f16 fragment (ISA 7.12.2): lanes 0-15 hold K 0-7 & 16-23,
// lanes 16-31 hold K 8-15 & 24-31 for row M = lane&15.
__device__ __forceinline__ v16h load_a_frag(const unsigned short* rowBase, int kOff) {
  v8h lo = *(const v8h*)(rowBase + kOff);        // K = kb .. kb+7
  v8h hi = *(const v8h*)(rowBase + kOff + 16);   // K = 16+kb .. 16+kb+7
  return __builtin_shufflevector(lo, hi, 0,1,2,3,4,5,6,7,8,9,10,11,12,13,14,15);
}

// B-matrix 32x16 f16 fragment: col N = lane&15; lanes 0-15 hold K 0-7 & 16-23,
// lanes 16-31 hold K 8-15 & 24-31 (same K order as A).
__device__ __forceinline__ v16h load_b_frag(const unsigned short* colBase, int kOff) {
  v8h lo = *(const v8h*)(colBase + kOff);
  v8h hi = *(const v8h*)(colBase + kOff + 16);
  return __builtin_shufflevector(lo, hi, 0,1,2,3,4,5,6,7,8,9,10,11,12,13,14,15);
}

__global__ __launch_bounds__(256)
void int4_gemm_wmma(const float* __restrict__ x,
                    const int*   __restrict__ Wp,
                    const float* __restrict__ scales,
                    const float* __restrict__ zeros,
                    const float* __restrict__ bias,
                    float*       __restrict__ out,
                    int M, int N, int K)
{
  // Double-buffered LDS tiles, f16 bit patterns in ushort storage.
  // As: [m][k] row-major; Bs: [n][k] (K-contiguous per output column).
  __shared__ unsigned short As[2][BM][LDA];
  __shared__ unsigned short Bs[2][BN][LDA];

  const int tid  = threadIdx.x;
  const int lane = tid & 31;
  const int wid  = tid >> 5;
  const int wm   = wid & 3;   // 4 m-bands of 32 rows
  const int wn   = wid >> 2;  // 2 n-bands of 64 cols

  const int m0 = blockIdx.y * BM;
  const int n0 = blockIdx.x * BN;
  const int KT = K / BK;

  // --- loader thread mapping ---
  const int xc4 = tid & 31;               // float4 index within an X row (32 * 4 = 128 K)
  const int xr0 = tid >> 5;               // X row start, stride 8 (8 rows * 16 passes = 128)
  const int bn  = tid & 127;              // local output column for W tile
  const int khb = (tid >> 7) * (BK / 4);  // packed-row base: 0 or 32 (of 64)

  auto load_tile = [&](int kt, int buf) {
    const int k0 = kt * BK;
    // X[128 x 128] f32 -> f16 (v_cvt_pk_rtz_f16_f32) -> As
#pragma unroll 4
    for (int p = 0; p < 16; ++p) {
      const int row = xr0 + p * 8;
      const float4 v = *(const float4*)(x + (size_t)(m0 + row) * K + k0 + xc4 * 4);
      unsigned* d = (unsigned*)&As[buf][row][xc4 * 4];
      d[0] = __builtin_bit_cast(unsigned, __builtin_amdgcn_cvt_pkrtz(v.x, v.y));
      d[1] = __builtin_bit_cast(unsigned, __builtin_amdgcn_cvt_pkrtz(v.z, v.w));
    }
    // W[64 x 128] packed int4 -> f16 pair via mantissa trick + v_pk_fma_f16.
    //   0x6400 | q  ==  f16(1024 + q)  exactly (q in 0..15)
    //   (q - zp)*sc == (1024+q)*sc + (-(1024+zp)*sc), fused => exact to f16.
    const int g = k0 / GS;
    const float sc = scales[(size_t)g * N + n0 + bn];
    const float zp = zeros [(size_t)g * N + n0 + bn];
    const _Float16 hs = (_Float16)sc;
    const _Float16 ho = (_Float16)(-(1024.0f + zp) * sc);
    const h2 hs2 = {hs, hs};
    const h2 ho2 = {ho, ho};
    const int* wp = Wp + (size_t)(k0 / 2 + khb) * N + n0 + bn;
    unsigned short* brow = &Bs[buf][bn][0];
#pragma unroll 8
    for (int j = 0; j < BK / 4; ++j) {
      const unsigned pv = (unsigned)wp[(size_t)j * N];
      const unsigned u  = (((pv << 12) | pv) & 0x000F000Fu) | 0x64006400u;
      const h2 hq = __builtin_bit_cast(h2, u);
      const h2 w  = hq * hs2 + ho2;   // v_pk_fma_f16 (contracted)
      *(unsigned*)&brow[2 * (khb + j)] = __builtin_bit_cast(unsigned, w);
    }
  };

  v8f acc[2][4] = {};

  load_tile(0, 0);
  __syncthreads();

  const int rsel = lane >> 4;
  const int rlo  = lane & 15;

  for (int kt = 0; kt < KT; ++kt) {
    const int cur = kt & 1;
    if (kt + 1 < KT) load_tile(kt + 1, cur ^ 1);  // prefetch next tile into other buffer

    const unsigned short* aRow0 = &As[cur][wm * 32 +      rlo][0];
    const unsigned short* aRow1 = &As[cur][wm * 32 + 16 + rlo][0];
    const unsigned short* bCol0 = &Bs[cur][wn * 64 +      rlo][0];
    const unsigned short* bCol1 = &Bs[cur][wn * 64 + 16 + rlo][0];
    const unsigned short* bCol2 = &Bs[cur][wn * 64 + 32 + rlo][0];
    const unsigned short* bCol3 = &Bs[cur][wn * 64 + 48 + rlo][0];

#pragma unroll
    for (int kk = 0; kk < BK / 32; ++kk) {
      const int aOff = kk * 32 + rsel * 8;
      const int bOff = kk * 32 + rsel * 8;
      v16h a0 = load_a_frag(aRow0, aOff);
      v16h a1 = load_a_frag(aRow1, aOff);
      v16h b0 = load_b_frag(bCol0, bOff);
      v16h b1 = load_b_frag(bCol1, bOff);
      v16h b2 = load_b_frag(bCol2, bOff);
      v16h b3 = load_b_frag(bCol3, bOff);
      acc[0][0] = __builtin_amdgcn_wmma_f32_16x16x32_f16(false, a0, false, b0, (short)0, acc[0][0], false, false);
      acc[1][0] = __builtin_amdgcn_wmma_f32_16x16x32_f16(false, a1, false, b0, (short)0, acc[1][0], false, false);
      acc[0][1] = __builtin_amdgcn_wmma_f32_16x16x32_f16(false, a0, false, b1, (short)0, acc[0][1], false, false);
      acc[1][1] = __builtin_amdgcn_wmma_f32_16x16x32_f16(false, a1, false, b1, (short)0, acc[1][1], false, false);
      acc[0][2] = __builtin_amdgcn_wmma_f32_16x16x32_f16(false, a0, false, b2, (short)0, acc[0][2], false, false);
      acc[1][2] = __builtin_amdgcn_wmma_f32_16x16x32_f16(false, a1, false, b2, (short)0, acc[1][2], false, false);
      acc[0][3] = __builtin_amdgcn_wmma_f32_16x16x32_f16(false, a0, false, b3, (short)0, acc[0][3], false, false);
      acc[1][3] = __builtin_amdgcn_wmma_f32_16x16x32_f16(false, a1, false, b3, (short)0, acc[1][3], false, false);
    }
    __syncthreads();
  }

  // Epilogue: C/D layout (ISA 7.12.2) — VGPR r holds M = r (+8 for lanes 16-31),
  // N = lane&15. Fused bias add, f32 out, non-temporal (write-once stream).
  const int rowBase = m0 + wm * 32 + (rsel << 3);
  const int colBase = n0 + wn * 64 + rlo;
#pragma unroll
  for (int nt = 0; nt < 4; ++nt) {
    const int col = colBase + nt * 16;
    const float bv = bias[col];
#pragma unroll
    for (int mt = 0; mt < 2; ++mt) {
      const int row0 = rowBase + mt * 16;
#pragma unroll
      for (int r = 0; r < 8; ++r) {
        __builtin_nontemporal_store(acc[mt][nt][r] + bv,
                                    out + (size_t)(row0 + r) * N + col);
      }
    }
  }
}

extern "C" void kernel_launch(void* const* d_in, const int* in_sizes, int n_in,
                              void* d_out, int out_size, void* d_ws, size_t ws_size,
                              hipStream_t stream) {
  const float* x      = (const float*)d_in[0];
  const int*   Wp     = (const int*)  d_in[1];
  const float* scales = (const float*)d_in[2];
  const float* zeros  = (const float*)d_in[3];
  const float* bias   = (const float*)d_in[4];
  float* out = (float*)d_out;

  const long long N = in_sizes[4];                        // bias: [N]
  const long long K = 2LL * ((long long)in_sizes[1] / N); // Wp: [K/2, N]
  const long long M = (long long)in_sizes[0] / K;         // x: [M, K]

  dim3 grid((unsigned)(N / BN), (unsigned)(M / BM));
  int4_gemm_wmma<<<grid, 256, 0, stream>>>(x, Wp, scales, zeros, bias, out,
                                           (int)M, (int)N, (int)K);
}
